// EfficientKANConv_68032281969066
// MI455X (gfx1250) — matrix operand order repaired and checked by reading, not compiled, not run on hardware
//
#include <hip/hip_runtime.h>
#include <hip/hip_bf16.h>

typedef __attribute__((ext_vector_type(16))) _Float16 v16h;
typedef __attribute__((ext_vector_type(8)))  _Float16 v8h;
typedef __attribute__((ext_vector_type(8)))  float    v8f;

#define O_CH   128
#define C_CH   32
#define HH     64
#define WW     64
#define F_K    9               // 3x3 patch
#define NBASE  8               // spline bases per feature
#define KPF    (1 + NBASE)     // 9 K-slots per (c,f)
#define KTOT   (C_CH * F_K * KPF)   // 2592 = 81 * 32
#define KSTEPS (KTOT / 32)          // 81
#define KPATCH (C_CH * F_K)         // 288 (patch-value K region)
#define NTILE  32
#define PPB    (HH * WW)            // 4096 positions per batch
#define NPOS   (8 * PPB)            // 32768

// d_ws layout: packed f16 weights (A fragments), then fp32 scaler[128]
#define WPACK_HALFS (KSTEPS * 8 * 32 * 16)       // 331776 halfs = 663552 B
#define SCALER_OFF_BYTES (WPACK_HALFS * 2)

// ---------------- kernel 1: per-out-channel spline scaler (mean over C*9) ----
__global__ void kan_scaler_kernel(const float* __restrict__ ss, float* __restrict__ scaler) {
    int o = threadIdx.x;
    if (o < O_CH) {
        float s = 0.f;
        for (int i = 0; i < KPATCH; ++i) s += ss[o * KPATCH + i];
        scaler[o] = s * (1.0f / (float)KPATCH);
    }
}

// ---------------- kernel 2: pack weights into WMMA A-fragment order ----------
// Layout: wpack[ks][mblk(8)][lane(32)][t(16)] f16.
// A-matrix 16x32 f16 per-lane K order (contiguous, matches B):
//   lane<16 : t=0..15 -> K=t       (K 0..15)
//   lane>=16: t=0..15 -> K=t+16    (K 16..31)
__global__ void kan_pack_kernel(const float* __restrict__ bw,
                                const float* __restrict__ sw,
                                const float* __restrict__ scaler,
                                _Float16* __restrict__ wpack) {
    int idx  = blockIdx.x * 256 + threadIdx.x;    // exact grid, no bounds check
    int t    = idx & 15;
    int lane = (idx >> 4) & 31;
    int mblk = (idx >> 9) & 7;
    int ks   = idx >> 12;
    int klocal = (lane < 16) ? t : t + 16;
    int k = ks * 32 + klocal;
    int M = mblk * 16 + (lane & 15);
    float val;
    if (k < KPATCH) {                       // base-conv weight region
        val = bw[M * KPATCH + k];           // k = c*9 + f
    } else {                                // spline region, scaler folded in
        int k2 = k - KPATCH;
        int cf = k2 >> 3, g = k2 & 7;
        val = scaler[M] * sw[(M * KPATCH + cf) * NBASE + g];
    }
    wpack[idx] = (_Float16)val;
}

// ---------------- kernel 3: fused basis-eval + GEMM --------------------------
// One WG = 128(M) x 32(N) output tile. LDS holds B-panel in fragment order:
// lds[ks][nb(2)][lane(32)][t(16)] f16.  B-matrix 32x16 per-lane order:
//   lane<16 : col N=lane,    t = K      (K 0..15)
//   lane>=16: col N=lane-16, t = K - 16 (K 16..31)
__global__ __launch_bounds__(256, 1)
void kan_main_kernel(const float* __restrict__ x,
                     const _Float16* __restrict__ wpack,
                     float* __restrict__ out) {
    extern __shared__ __align__(32) _Float16 lds[];   // 81*2*32*16 = 82944 halfs (162 KB)

    const int tid = threadIdx.x;
    const int n0  = blockIdx.x * NTILE;
    const int b   = n0 >> 12;                 // NTILE divides PPB -> one batch
    const int p0  = n0 & (PPB - 1);

    // ---- Phase A: compute patch values + cubic B-spline bases into LDS ----
    for (int e = tid; e < KPATCH * NTILE; e += 256) {
        int nl = e & 31;                      // position within tile (lane-contiguous)
        int cf = e >> 5;                      // (c,f) index 0..287
        int c  = cf / 9, f = cf - c * 9;
        int dy = f / 3,  dx = f - dy * 3;
        int p  = p0 + nl;
        int yy = p >> 6, xx = p & 63;
        int iy = yy + dy - 1, ix = xx + dx - 1;
        float val = 0.f;
        if (iy >= 0 && iy < HH && ix >= 0 && ix < WW)
            val = x[((b * C_CH + c) * HH + iy) * WW + ix];

        int nb = nl >> 4, ncol = nl & 15;

        // patch value at K = cf (single f16 store)
        {
            int ks = cf >> 5, kl = cf & 31;
            int lane2 = ((kl >> 4) << 4) + ncol;
            lds[(((ks * 2) + nb) * 32 + lane2) * 16 + (kl & 15)] = (_Float16)val;
        }

        // uniform cubic B-spline: grid knots t_i = -2.2 + 0.4*i, i=0..11
        // only 4 bases nonzero; interval i = floor((u+2.2)/0.4), s in [0,1)
        float hh[8] = {0.f,0.f,0.f,0.f,0.f,0.f,0.f,0.f};
        float u  = (val + 2.2f) * 2.5f;
        float fi = floorf(u);
        int   i  = (int)fi;
        if (i >= 0 && i <= 10) {
            float s  = u - fi;
            float s1 = 1.f - s;
            float s2 = s * s, s3 = s2 * s;
            float b0 = s1 * s1 * s1 * (1.f / 6.f);
            float b1 = (3.f * s3 - 6.f * s2 + 4.f) * (1.f / 6.f);
            float b2 = (-3.f * s3 + 3.f * s2 + 3.f * s + 1.f) * (1.f / 6.f);
            float b3 = s3 * (1.f / 6.f);
#pragma unroll
            for (int j = 0; j < 8; ++j) {
                int r = j - (i - 3);
                if (r >= 0 && r < 4)
                    hh[j] = (r == 0) ? b0 : (r == 1) ? b1 : (r == 2) ? b2 : b3;
            }
        }
        v8h hv;
#pragma unroll
        for (int j = 0; j < 8; ++j) hv[j] = (_Float16)hh[j];

        // 8 bases occupy 8 contiguous K slots starting at 288 + cf*8 (8-aligned)
        int kb  = KPATCH + (cf << 3);
        int ks2 = kb >> 5, kl2 = kb & 31;               // kl2 in {0,8,16,24}
        int lane2 = ((kl2 >> 4) << 4) + ncol;
        int tst   = kl2 & 15;                           // 0 or 8 -> 16B aligned
        *(v8h*)&lds[(((ks2 * 2) + nb) * 32 + lane2) * 16 + tst] = hv;
    }
    __syncthreads();

    // ---- Phase B: 81-step WMMA GEMM, full M=128 across 8 waves -------------
    const int wave = tid >> 5;
    const int lane = tid & 31;
    v8f acc0 = {};
    v8f acc1 = {};
    const _Float16* wp = wpack + ((size_t)(wave * 32 + lane)) * 16;   // 32B-aligned

#pragma unroll 3
    for (int ks = 0; ks < KSTEPS; ++ks) {
        v16h a  = *(const v16h*)(wp + (size_t)ks * (8 * 32 * 16));
        v16h b0 = *(const v16h*)(&lds[((ks * 2 + 0) * 32 + lane) * 16]);
        v16h b1 = *(const v16h*)(&lds[((ks * 2 + 1) * 32 + lane) * 16]);
        acc0 = __builtin_amdgcn_wmma_f32_16x16x32_f16(false, a, false, b0,
                                                      (short)0, acc0, false, false);
        acc1 = __builtin_amdgcn_wmma_f32_16x16x32_f16(false, a, false, b1,
                                                      (short)0, acc1, false, false);
    }

    // ---- store: D layout -> M = r + 8*(lane>=16), N = lane&15 --------------
    const int mhi  = (lane >> 4) << 3;
    const int ncol = lane & 15;
    float* obase = out + (size_t)b * O_CH * PPB + p0;
#pragma unroll
    for (int r = 0; r < 8; ++r) {
        int o = wave * 16 + r + mhi;
        obase[(size_t)o * PPB + ncol]      = acc0[r];
        obase[(size_t)o * PPB + 16 + ncol] = acc1[r];
    }
}

extern "C" void kernel_launch(void* const* d_in, const int* in_sizes, int n_in,
                              void* d_out, int out_size, void* d_ws, size_t ws_size,
                              hipStream_t stream) {
    const float* x  = (const float*)d_in[0];   // (8,32,64,64)
    const float* bw = (const float*)d_in[1];   // (128,32,3,3)
    const float* sw = (const float*)d_in[2];   // (128,32,3,3,8)
    const float* ss = (const float*)d_in[3];   // (128,32,3,3)
    float* out = (float*)d_out;                // (8,128,64,64)

    _Float16* wpack  = (_Float16*)d_ws;
    float*    scaler = (float*)((char*)d_ws + SCALER_OFF_BYTES);

    kan_scaler_kernel<<<1, 128, 0, stream>>>(ss, scaler);
    kan_pack_kernel<<<WPACK_HALFS / 256, 256, 0, stream>>>(bw, sw, scaler, wpack);

    size_t shbytes = (size_t)KSTEPS * 2 * 32 * 16 * sizeof(_Float16);  // 165888 B
    kan_main_kernel<<<NPOS / NTILE, 256, shbytes, stream>>>(x, wpack, out);
}
